// highway_conv1_1_23897198035644
// MI455X (gfx1250) — matrix operand order repaired and checked by reading, not compiled, not run on hardware
//
#include <hip/hip_runtime.h>
#include <hip/hip_bf16.h>
#include <math.h>

// ---------------------------------------------------------------------------
// CDNA5 (gfx1250) WMMA types
// ---------------------------------------------------------------------------
typedef __attribute__((ext_vector_type(16))) __bf16 v16bf;
typedef __attribute__((ext_vector_type(8)))  __bf16 v8bf;
typedef __attribute__((ext_vector_type(8)))  float  v8f;
typedef __attribute__((ext_vector_type(4)))  int    v4i;

#define B_    32
#define Hh    28
#define Ww    28
#define NPIX  784      // 28*28
#define C_    384
#define HID_  1536
#define EPS_  1e-5f

// K1 LDS tile: 16 rows x 384 k (bf16), row stride padded to 392 elems (784 B)
#define K1_ROWB   784
#define K1_TILEE  (16 * 392)
// K3 LDS tile: 64 rows x 64 k (bf16), row stride padded to 72 elems (144 B)
#define K3_ROWB   144
#define K3_TILEE  (64 * 72)

// ---------------------------------------------------------------------------
// CDNA5 async global->LDS copies (no VGPR data path, tracked by ASYNCcnt).
// Falls back to regs+ds_store staging if the builtin is not declared.
// ---------------------------------------------------------------------------
#if defined(__has_builtin)
#if __has_builtin(__builtin_amdgcn_global_load_async_to_lds_b128)
#define USE_ASYNC_LDS 1
#endif
#endif

__device__ __forceinline__ void async_copy16(const void* g, void* l) {
#ifdef USE_ASYNC_LDS
  __builtin_amdgcn_global_load_async_to_lds_b128(
      (__attribute__((address_space(1))) v4i*)g,
      (__attribute__((address_space(3))) v4i*)l, 0, 0);
#endif
}
__device__ __forceinline__ void async_wait0() {
#ifdef USE_ASYNC_LDS
  asm volatile("s_wait_asynccnt 0x0" ::: "memory");
#endif
}

// ---------------------------------------------------------------------------
// Fragment helpers (contiguous k order, same for A and B).
// A 16x32 (MxK): lane l -> row m=l&15, hi=l>>4; elem e: k = 16*hi+e
// B 32x16 (KxN): lane l -> col n=l&15, hi=l>>4; elem e: k = 16*hi+e
// ---------------------------------------------------------------------------
__device__ __forceinline__ v16bf make_a(const v8bf lo, const v8bf hh) {
  v16bf a;
#pragma unroll
  for (int e = 0; e < 8; ++e) { a[e] = lo[e]; a[e + 8] = hh[e]; }
  return a;
}

__device__ __forceinline__ v16bf lds_a_frag(const __bf16* __restrict__ tile,
                                            int rowB, int row, int hi, int ks) {
  const char* base = (const char*)tile + row * rowB + hi * 32 + ks * 64;
  return make_a(*(const v8bf*)base, *(const v8bf*)(base + 16));
}

__device__ __forceinline__ v16bf load_b_frag(const __bf16* __restrict__ col_k0,
                                             int hi) {
  v16bf b;
  const __bf16* p = col_k0 + 16 * hi;
#pragma unroll
  for (int e = 0; e < 16; ++e) b[e] = p[e];
  return b;
}

// Branchless exact-GELU: erf via Abramowitz-Stegun 7.1.26 (|err| < 1.5e-7).
__device__ __forceinline__ float gelu_exact(float v) {
  const float ax = fabsf(v) * 0.70710678118654752440f;
  const float t  = __frcp_rn(1.0f + 0.3275911f * ax);
  const float poly =
      t * (0.254829592f +
      t * (-0.284496736f +
      t * (1.421413741f +
      t * (-1.453152027f + t * 1.061405429f))));
  float erfv = 1.0f - poly * __expf(-ax * ax);
  erfv = copysignf(erfv, v);
  return 0.5f * v * (1.0f + erfv);
}

// ---------------------------------------------------------------------------
// Prep: streaming f32 -> bf16 convert, 4 elements/thread (n % 1024 == 0).
// ---------------------------------------------------------------------------
__global__ void __launch_bounds__(256)
k_cvt_bf16(const float* __restrict__ src, __bf16* __restrict__ dst) {
  const int i = (blockIdx.x * 256 + threadIdx.x) * 4;
  const float4 f = *(const float4*)(src + i);
  __bf16 o[4] = {(__bf16)f.x, (__bf16)f.y, (__bf16)f.z, (__bf16)f.w};
  *(uint2*)(dst + i) = *(const uint2*)o;
}

// ---------------------------------------------------------------------------
// Kernel 1: h1[b][oc][n] = BN1(GELU(x[b][n][:] @ w1[oc][:] + b1[oc]))  (bf16)
// grid (HID/128, B), block 256 (8 waves, each owning a 16-wide oc tile).
// x M-tiles are shared by all 8 waves via double-buffered LDS, filled by
// async global->LDS copies issued before the WMMA chain of the previous
// tile (true compute/copy overlap, zero staging registers).
// ---------------------------------------------------------------------------
__global__ void __launch_bounds__(256)
k_gemm1(const __bf16* __restrict__ xbf, const __bf16* __restrict__ w1bf,
        const float* __restrict__ b1, const float* __restrict__ g1,
        const float* __restrict__ be1, const float* __restrict__ m1,
        const float* __restrict__ v1, __bf16* __restrict__ h1) {
  __shared__ __bf16 sA[2][K1_TILEE];

  const int tid  = threadIdx.x;
  const int bidx = blockIdx.y;
  const int wave = tid >> 5;
  const int lane = tid & 31;
  const int hi   = lane >> 4;
  const int row  = lane & 15;
  const int col  = blockIdx.x * 128 + wave * 16 + row;   // output channel

  v16bf bw[12];
#pragma unroll
  for (int ks = 0; ks < 12; ++ks)
    bw[ks] = load_b_frag(w1bf + (size_t)col * C_ + ks * 32, hi);

  const float s1  = g1[col] * rsqrtf(v1[col] + EPS_);
  const float sh1 = be1[col] - m1[col] * s1;
  const float b1c = b1[col];

  const __bf16* xb  = xbf + (size_t)bidx * NPIX * C_;
  __bf16*       h1c = h1 + ((size_t)bidx * HID_ + col) * NPIX;

  // Staging map: 16x384 tile = 12288 contiguous bytes = 768 16B chunks,
  // 3 per thread; LDS rows padded to 784 B (conflict-free fragment reads).
#ifdef USE_ASYNC_LDS
#pragma unroll
  for (int i = 0; i < 3; ++i) {
    const int c = tid + 256 * i;
    async_copy16((const char*)xb + c * 16,
                 (char*)sA[0] + (c / 48) * K1_ROWB + (c % 48) * 16);
  }
  async_wait0();
#else
  uint4 R[3];
#pragma unroll
  for (int i = 0; i < 3; ++i)
    R[i] = *(const uint4*)((const char*)xb + (tid + 256 * i) * 16);
#pragma unroll
  for (int i = 0; i < 3; ++i) {
    const int c = tid + 256 * i;
    *(uint4*)((char*)sA[0] + (c / 48) * K1_ROWB + (c % 48) * 16) = R[i];
  }
#endif
  __syncthreads();

  for (int mt = 0; mt < 49; ++mt) {
    const __bf16* cur = sA[mt & 1];
    __bf16*       nxt = sA[(mt + 1) & 1];

    if (mt < 48) {
      const char* src = (const char*)(xb + (size_t)(mt + 1) * 16 * C_);
#ifdef USE_ASYNC_LDS
#pragma unroll
      for (int i = 0; i < 3; ++i) {
        const int c = tid + 256 * i;
        async_copy16(src + c * 16,
                     (char*)nxt + (c / 48) * K1_ROWB + (c % 48) * 16);
      }
#else
#pragma unroll
      for (int i = 0; i < 3; ++i)
        R[i] = *(const uint4*)(src + (tid + 256 * i) * 16);
#endif
    }

    v8f acc = {};
#pragma unroll
    for (int ks = 0; ks < 12; ++ks) {
      v16bf a = lds_a_frag(cur, K1_ROWB, row, hi, ks);
      acc = __builtin_amdgcn_wmma_f32_16x16x32_bf16(
          false, a, false, bw[ks], (short)0, acc, false, false);
    }

#ifndef USE_ASYNC_LDS
    if (mt < 48) {
#pragma unroll
      for (int i = 0; i < 3; ++i) {
        const int c = tid + 256 * i;
        *(uint4*)((char*)nxt + (c / 48) * K1_ROWB + (c % 48) * 16) = R[i];
      }
    }
#endif

    // n = mt*16 + 8*hi + v, v=0..7 -> one contiguous 16B store per lane.
    __bf16 o[8];
#pragma unroll
    for (int v = 0; v < 8; ++v) {
      float val = acc[v] + b1c;
      val       = gelu_exact(val);
      o[v]      = (__bf16)(val * s1 + sh1);
    }
    *(uint4*)(h1c + mt * 16 + 8 * hi) = *(const uint4*)o;

    async_wait0();
    __syncthreads();
  }
}

// ---------------------------------------------------------------------------
// Kernel 2: depthwise 3x3 (pad 1) + bd + GELU + BN2.
// h1[b][oc][28*28] (bf16) -> h2[(b*784+n)][oc] (bf16, K-contiguous for GEMM2).
// ---------------------------------------------------------------------------
__global__ void __launch_bounds__(256)
k_dwconv(const __bf16* __restrict__ h1, const float* __restrict__ wd,
         const float* __restrict__ bd, const float* __restrict__ g2,
         const float* __restrict__ be2, const float* __restrict__ m2,
         const float* __restrict__ v2, __bf16* __restrict__ h2) {
  const int idx = blockIdx.x * 256 + threadIdx.x;   // < B_*HID_*NPIX (exact)
  const int n   = idx % NPIX;
  const int t   = idx / NPIX;
  const int oc  = t % HID_;
  const int b   = t / HID_;
  const int yy  = n / Ww;
  const int xx  = n % Ww;

  const __bf16* plane = h1 + ((size_t)b * HID_ + oc) * NPIX;
  const float*  wk    = wd + (size_t)oc * 9;

  float acc = 0.0f;
#pragma unroll
  for (int dy = -1; dy <= 1; ++dy) {
#pragma unroll
    for (int dx = -1; dx <= 1; ++dx) {
      const int Y = yy + dy, X = xx + dx;
      if (Y >= 0 && Y < Hh && X >= 0 && X < Ww)
        acc += (float)plane[Y * Ww + X] * wk[(dy + 1) * 3 + (dx + 1)];
    }
  }
  float val = acc + bd[oc];
  val       = gelu_exact(val);
  const float s2 = g2[oc] * rsqrtf(v2[oc] + EPS_);
  val = val * s2 + (be2[oc] - m2[oc] * s2);
  h2[(size_t)(b * NPIX + n) * HID_ + oc] = (__bf16)val;
}

// ---------------------------------------------------------------------------
// Kernel 3: out[m][c] = BN3(h2[m][:] @ w2[c][:] + b2[c]) + x[m][c]
// M = 25088 (flat), K = 1536, N = 384.
// grid (C/128, M/64), block 256 (8 waves sharing one 64x64 A k-chunk in
// double-buffered LDS, filled by async copies); per-wave B-fragments are
// double-buffered in registers. 24 iterations x 8 WMMAs.
// ---------------------------------------------------------------------------
__global__ void __launch_bounds__(256)
k_gemm2(const __bf16* __restrict__ h2, const __bf16* __restrict__ w2bf,
        const float* __restrict__ b2, const float* __restrict__ g3,
        const float* __restrict__ be3, const float* __restrict__ m3,
        const float* __restrict__ v3, const float* __restrict__ x,
        float* __restrict__ out) {
  __shared__ __bf16 sA[2][K3_TILEE];

  const int tid  = threadIdx.x;
  const int wave = tid >> 5;
  const int lane = tid & 31;
  const int hi   = lane >> 4;
  const int col  = blockIdx.x * 128 + wave * 16 + (lane & 15); // c in [0,384)
  const int m0   = blockIdx.y * 64;                            // 4 row tiles

  const float s3  = g3[col] * rsqrtf(v3[col] + EPS_);
  const float sh3 = be3[col] - m3[col] * s3;
  const float b2c = b2[col];

  const __bf16* abase = h2 + (size_t)m0 * HID_;
  const __bf16* wcol  = w2bf + (size_t)col * HID_;

  // Stage A chunk [64 rows x 64 k]: 512 16B chunks, 2 per thread.
  // Global: row stride 3072 B; LDS: row stride 144 B (conflict-free).
#ifdef USE_ASYNC_LDS
#pragma unroll
  for (int i = 0; i < 2; ++i) {
    const int c = tid + 256 * i;
    async_copy16((const char*)abase + (c / 8) * (HID_ * 2) + (c % 8) * 16,
                 (char*)sA[0] + (c / 8) * K3_ROWB + (c % 8) * 16);
  }
  async_wait0();
#else
  uint4 R[2];
#pragma unroll
  for (int i = 0; i < 2; ++i) {
    const int c = tid + 256 * i;
    R[i] = *(const uint4*)((const char*)abase + (c / 8) * (HID_ * 2) + (c % 8) * 16);
  }
#pragma unroll
  for (int i = 0; i < 2; ++i) {
    const int c = tid + 256 * i;
    *(uint4*)((char*)sA[0] + (c / 8) * K3_ROWB + (c % 8) * 16) = R[i];
  }
#endif
  v16bf b0 = load_b_frag(wcol, hi);
  v16bf b1 = load_b_frag(wcol + 32, hi);
  __syncthreads();

  v8f acc[4] = {};
  for (int g = 0; g < 24; ++g) {
    const __bf16* cur = sA[g & 1];
    __bf16*       nxt = sA[(g + 1) & 1];

    v16bf nb0, nb1;
    if (g < 23) {
      const int k0n = (g + 1) * 64;
      const char* src = (const char*)abase + k0n * 2;
#ifdef USE_ASYNC_LDS
#pragma unroll
      for (int i = 0; i < 2; ++i) {
        const int c = tid + 256 * i;
        async_copy16(src + (c / 8) * (HID_ * 2) + (c % 8) * 16,
                     (char*)nxt + (c / 8) * K3_ROWB + (c % 8) * 16);
      }
#else
#pragma unroll
      for (int i = 0; i < 2; ++i) {
        const int c = tid + 256 * i;
        R[i] = *(const uint4*)(src + (c / 8) * (HID_ * 2) + (c % 8) * 16);
      }
#endif
      nb0 = load_b_frag(wcol + k0n, hi);
      nb1 = load_b_frag(wcol + k0n + 32, hi);
    }

#pragma unroll
    for (int t = 0; t < 4; ++t) {
      v16bf a = lds_a_frag(cur, K3_ROWB, 16 * t + (lane & 15), hi, 0);
      acc[t] = __builtin_amdgcn_wmma_f32_16x16x32_bf16(
          false, a, false, b0, (short)0, acc[t], false, false);
    }
#pragma unroll
    for (int t = 0; t < 4; ++t) {
      v16bf a = lds_a_frag(cur, K3_ROWB, 16 * t + (lane & 15), hi, 1);
      acc[t] = __builtin_amdgcn_wmma_f32_16x16x32_bf16(
          false, a, false, b1, (short)0, acc[t], false, false);
    }

    if (g < 23) {
#ifndef USE_ASYNC_LDS
#pragma unroll
      for (int i = 0; i < 2; ++i) {
        const int c = tid + 256 * i;
        *(uint4*)((char*)nxt + (c / 8) * K3_ROWB + (c % 8) * 16) = R[i];
      }
#endif
      b0 = nb0;
      b1 = nb1;
    }
    async_wait0();
    __syncthreads();
  }

#pragma unroll
  for (int t = 0; t < 4; ++t) {
#pragma unroll
    for (int v = 0; v < 8; ++v) {
      const int m = m0 + 16 * t + v + 8 * hi;
      float val   = acc[t][v] + b2c;
      val         = val * s3 + sh3;
      val        += x[(size_t)m * C_ + col];
      out[(size_t)m * C_ + col] = val;
    }
  }
}

// ---------------------------------------------------------------------------
// Launcher
// ---------------------------------------------------------------------------
extern "C" void kernel_launch(void* const* d_in, const int* in_sizes, int n_in,
                              void* d_out, int out_size, void* d_ws, size_t ws_size,
                              hipStream_t stream) {
  const float* x   = (const float*)d_in[0];
  const float* w1  = (const float*)d_in[1];
  const float* b1  = (const float*)d_in[2];
  const float* g1  = (const float*)d_in[3];
  const float* be1 = (const float*)d_in[4];
  const float* m1  = (const float*)d_in[5];
  const float* v1  = (const float*)d_in[6];
  const float* wd  = (const float*)d_in[7];
  const float* bd  = (const float*)d_in[8];
  const float* g2  = (const float*)d_in[9];
  const float* be2 = (const float*)d_in[10];
  const float* m2  = (const float*)d_in[11];
  const float* v2  = (const float*)d_in[12];
  const float* w2  = (const float*)d_in[13];
  const float* b2  = (const float*)d_in[14];
  const float* g3  = (const float*)d_in[15];
  const float* be3 = (const float*)d_in[16];
  const float* m3  = (const float*)d_in[17];
  const float* v3  = (const float*)d_in[18];
  float*       out = (float*)d_out;

  // Workspace (bf16), all fully rewritten before each read (deterministic):
  //   h1: 38,535,168 | h2: 38,535,168 | xbf: 9,633,792 | w1bf/w2bf: 589,824 ea.
  const size_t elems = (size_t)B_ * HID_ * NPIX;
  const size_t xel   = (size_t)B_ * NPIX * C_;
  const size_t wel   = (size_t)HID_ * C_;
  __bf16* h1   = (__bf16*)d_ws;
  __bf16* h2   = h1 + elems;
  __bf16* xbf  = h2 + elems;
  __bf16* w1bf = xbf + xel;
  __bf16* w2bf = w1bf + wel;

  // P0: one-shot f32 -> bf16 conversions (all sizes divisible by 1024)
  k_cvt_bf16<<<(unsigned)(xel / 1024), 256, 0, stream>>>(x, xbf);
  k_cvt_bf16<<<(unsigned)(wel / 1024), 256, 0, stream>>>(w1, w1bf);
  k_cvt_bf16<<<(unsigned)(wel / 1024), 256, 0, stream>>>(w2, w2bf);

  // K1: 1x1 conv C->HID + GELU + BN1
  k_gemm1<<<dim3(HID_ / 128, B_), 256, 0, stream>>>(xbf, w1bf, b1, g1, be1, m1, v1, h1);

  // K2: depthwise 3x3 + GELU + BN2 (38.5M outputs, 1 per thread)
  k_dwconv<<<(unsigned)(elems / 256), 256, 0, stream>>>(h1, wd, bd, g2, be2, m2, v2, h2);

  // K3: 1x1 conv HID->C + BN3 + residual
  k_gemm2<<<dim3(C_ / 128, (B_ * NPIX) / 64), 256, 0, stream>>>(
      h2, w2bf, b2, g3, be3, m3, v3, x, out);
}
